// GCN_20461224198523
// MI455X (gfx1250) — matrix operand order repaired and checked by reading, not compiled, not run on hardware
//
#include <hip/hip_runtime.h>
#include <math.h>

// ---------------------------------------------------------------------------
// GCN 3-layer forward for gfx1250 (MI455X). Dense GEMMs use
// v_wmma_f32_16x16x32_bf16 (wave32, f32 accum), register-blocked 4x along M
// with named accumulators (no dynamic VGPR indexing) and a barrier-free
// inner K loop. Sparse aggregation uses hardware fp32 atomics.
// ---------------------------------------------------------------------------

typedef __attribute__((ext_vector_type(16))) __bf16 v16bf;
typedef __attribute__((ext_vector_type(8)))  float  v8f;

#define NN 50000
#define NE 800000
#define FIN 512
#define FH  256
#define FOUT 128

__device__ __forceinline__ unsigned short f32_to_bf16_rne(float f) {
    unsigned int u = __float_as_uint(f);
    unsigned int r = u + 0x7FFFu + ((u >> 16) & 1u);
    return (unsigned short)(r >> 16);
}

// ---------------- elementwise / setup kernels ------------------------------

__global__ void k_zero_f32(float* __restrict__ p, int n) {
    int i = blockIdx.x * blockDim.x + threadIdx.x;
    if (i < n) p[i] = 0.0f;
}

__global__ void k_cvt_bf16(const float* __restrict__ in, unsigned short* __restrict__ out, int n) {
    int i = blockIdx.x * blockDim.x + threadIdx.x;
    if (i < n) out[i] = f32_to_bf16_rne(in[i]);
}

__global__ void k_deg_accum(const int* __restrict__ dst, float* __restrict__ deg, int e) {
    int i = blockIdx.x * blockDim.x + threadIdx.x;
    if (i < e) unsafeAtomicAdd(&deg[dst[i]], 1.0f);
}

__global__ void k_dinv(float* __restrict__ deg, int n) {
    int i = blockIdx.x * blockDim.x + threadIdx.x;
    if (i < n) deg[i] = rsqrtf(deg[i] + 1.0f);   // in-place: deg -> dinv
}

// ---------------- WMMA bf16 GEMM:  C[M,F] = A[M,K] @ B[K,F] (+bias) --------
//
// Block: 256 threads = 8 waves. Block tile = 64 rows x 128 cols.
// Each wave owns one 16-col tile and 4 row-tiles (named v8f accumulators).
// B panel (128 cols x 256 K) staged per K-chunk in LDS, transposed to
// K-contiguous per column, so each lane's B fragment is two ds_load_b128.
// A fragments are read straight from row-major global memory as two
// global_load_b128 per row-tile (A layout: row = lane&15, lane-half selects
// K 0..7/16..23 vs 8..15/24..31 -> contiguous 16B + 16B at +32B).
// Ragged M edge: clamp row bases for compute, guard only the stores.
// Inner K loop: 2 ds_load_b128 + 8 global_load_b128 + 4 wmma, no barriers.

template<int TK>
__global__ __launch_bounds__(256) void k_gemm_bf16(
    const unsigned short* __restrict__ A,   // M x TK (bf16 bits)
    const unsigned short* __restrict__ B,   // TK x F (bf16 bits)
    const float* __restrict__ bias,         // F or nullptr
    float* __restrict__ C,                  // M x F
    int M, int F)
{
    __shared__ __align__(16) unsigned short lsBt[128 * 256];   // 64 KB: [col][k]

    const int tid  = threadIdx.x;
    const int w    = tid >> 5;
    const int lane = tid & 31;
    const int nn   = lane & 15;       // col within tile / A fragment row
    const int hh   = lane >> 4;       // lane-half select
    const int c0   = blockIdx.x * 128;
    const int row0 = blockIdx.y * 64;

    // clamped row bases for compute (duplicates on the ragged last block)
    const int r0 = row0;
    const int r1 = min(row0 + 16, M - 16);
    const int r2 = min(row0 + 32, M - 16);
    const int r3 = min(row0 + 48, M - 16);

    // per-row-tile A fragment base pointers (lane-half K offset baked in)
    const unsigned short* pa0 = A + (size_t)(r0 + nn) * TK + hh * 8;
    const unsigned short* pa1 = A + (size_t)(r1 + nn) * TK + hh * 8;
    const unsigned short* pa2 = A + (size_t)(r2 + nn) * TK + hh * 8;
    const unsigned short* pa3 = A + (size_t)(r3 + nn) * TK + hh * 8;

    const unsigned short* pbB = &lsBt[(w * 16 + nn) * 256 + hh * 8];

    v8f acc0{}, acc1{}, acc2{}, acc3{};

    for (int kc = 0; kc < TK; kc += 256) {
        // ---- stage B chunk transposed: 256 k x 128 cols -> lsBt[col][k] ----
        if (kc) __syncthreads();
        #pragma unroll 4
        for (int p = tid; p < 256 * 64; p += 256) {   // 16384 u32 pairs
            int kk = p >> 6;                          // 0..255
            int f  = (p & 63) * 2;                    // 0..126 even
            unsigned int v = *(const unsigned int*)&B[(kc + kk) * F + (c0 + f)];
            lsBt[f * 256 + kk]       = (unsigned short)(v & 0xFFFFu);
            lsBt[(f + 1) * 256 + kk] = (unsigned short)(v >> 16);
        }
        __syncthreads();

        // ---- barrier-free MAC loop over this 256-K chunk ----
        #pragma unroll 2
        for (int k0 = 0; k0 < 256; k0 += 32) {
            union { uint4 u[2]; v16bf v; } ub;
            ub.u[0] = *(const uint4*)(pbB + k0);
            ub.u[1] = *(const uint4*)(pbB + k0 + 16);

            union { uint4 u[2]; v16bf v; } ua0, ua1, ua2, ua3;
            const int ko = kc + k0;
            ua0.u[0] = *(const uint4*)(pa0 + ko);
            ua0.u[1] = *(const uint4*)(pa0 + ko + 16);
            ua1.u[0] = *(const uint4*)(pa1 + ko);
            ua1.u[1] = *(const uint4*)(pa1 + ko + 16);
            ua2.u[0] = *(const uint4*)(pa2 + ko);
            ua2.u[1] = *(const uint4*)(pa2 + ko + 16);
            ua3.u[0] = *(const uint4*)(pa3 + ko);
            ua3.u[1] = *(const uint4*)(pa3 + ko + 16);

            acc0 = __builtin_amdgcn_wmma_f32_16x16x32_bf16(
                       false, ua0.v, false, ub.v, (short)0, acc0, false, false);
            acc1 = __builtin_amdgcn_wmma_f32_16x16x32_bf16(
                       false, ua1.v, false, ub.v, (short)0, acc1, false, false);
            acc2 = __builtin_amdgcn_wmma_f32_16x16x32_bf16(
                       false, ua2.v, false, ub.v, (short)0, acc2, false, false);
            acc3 = __builtin_amdgcn_wmma_f32_16x16x32_bf16(
                       false, ua3.v, false, ub.v, (short)0, acc3, false, false);
        }
    }

    // ---- epilogue: C/D layout -> M = hh*8 + v, N = nn (32-bit index math) --
    const int col = c0 + w * 16 + nn;
    const float bv = bias ? bias[col] : 0.0f;
    const int mbase = hh * 8;

    {   // row-tile 0 (always valid)
        #pragma unroll
        for (int v = 0; v < 8; ++v)
            C[(row0 + mbase + v) * F + col] = acc0[v] + bv;
    }
    if (row0 + 16 < M) {
        #pragma unroll
        for (int v = 0; v < 8; ++v)
            C[(row0 + 16 + mbase + v) * F + col] = acc1[v] + bv;
    }
    if (row0 + 32 < M) {
        #pragma unroll
        for (int v = 0; v < 8; ++v)
            C[(row0 + 32 + mbase + v) * F + col] = acc2[v] + bv;
    }
    if (row0 + 48 < M) {
        #pragma unroll
        for (int v = 0; v < 8; ++v)
            C[(row0 + 48 + mbase + v) * F + col] = acc3[v] + bv;
    }
}

// ---------------- GCN aggregation ------------------------------------------

// agg[i,f] = h[i,f] * dinv[i]^2 + b[f]     (self-loop + bias init)
__global__ void k_self_init(const float* __restrict__ h, const float* __restrict__ dinv,
                            const float* __restrict__ b, float* __restrict__ agg,
                            int F, int total) {
    int i = blockIdx.x * blockDim.x + threadIdx.x;
    if (i < total) {
        int node = i / F;
        int f    = i - node * F;
        float dv = dinv[node];
        agg[i] = h[i] * dv * dv + b[f];
    }
}

// out[i,f] += h[i,f] * dinv[i]^2 + b[f]    (layer-3 variant, accumulates)
__global__ void k_add_self(const float* __restrict__ h, const float* __restrict__ dinv,
                           const float* __restrict__ b, float* __restrict__ out,
                           int F, int total) {
    int i = blockIdx.x * blockDim.x + threadIdx.x;
    if (i < total) {
        int node = i / F;
        int f    = i - node * F;
        float dv = dinv[node];
        out[i] += h[i] * dv * dv + b[f];
    }
}

// one edge per block; threads span features (coalesced gather + fp32 atomics)
__global__ void k_scatter(const float* __restrict__ h, const int* __restrict__ src,
                          const int* __restrict__ dst, const float* __restrict__ dinv,
                          float* __restrict__ agg, int F) {
    int e = blockIdx.x;
    int s = src[e];
    int d = dst[e];
    float norm = dinv[s] * dinv[d];
    int f = threadIdx.x;
    unsafeAtomicAdd(&agg[d * F + f], h[s * F + f] * norm);
}

// ---------------- BatchNorm (two-pass, coalesced) --------------------------

__global__ __launch_bounds__(256) void k_bn_partial(
    const float* __restrict__ x, float* __restrict__ sum, float* __restrict__ sumsq,
    int n, int F, int rows_per_block) {
    int f  = threadIdx.x;                      // blockDim.x == F == 256
    int r0 = blockIdx.x * rows_per_block;
    int r1 = r0 + rows_per_block; if (r1 > n) r1 = n;
    float s = 0.0f, q = 0.0f;
    for (int r = r0; r < r1; ++r) {
        float v = x[r * F + f];
        s += v; q += v * v;
    }
    unsafeAtomicAdd(&sum[f], s);
    unsafeAtomicAdd(&sumsq[f], q);
}

__global__ void k_bn_final(const float* __restrict__ sum, const float* __restrict__ sumsq,
                           float* __restrict__ mean, float* __restrict__ rstd, int n, int F) {
    int f = blockIdx.x * blockDim.x + threadIdx.x;
    if (f < F) {
        float m   = sum[f] / (float)n;
        float var = sumsq[f] / (float)n - m * m;
        mean[f] = m;
        rstd[f] = rsqrtf(var + 1e-5f);
    }
}

// BN apply + ELU + convert to bf16 (feeds next GEMM directly)
__global__ void k_bn_elu_bf16(const float* __restrict__ x, const float* __restrict__ mean,
                              const float* __restrict__ rstd, const float* __restrict__ gamma,
                              const float* __restrict__ beta, unsigned short* __restrict__ out,
                              int F, int total) {
    int i = blockIdx.x * blockDim.x + threadIdx.x;
    if (i < total) {
        int f = i % F;
        float v = (x[i] - mean[f]) * rstd[f] * gamma[f] + beta[f];
        v = (v > 0.0f) ? v : expm1f(v);
        out[i] = f32_to_bf16_rne(v);
    }
}

// ---------------- row-wise log_softmax (F = 128, in-place) -----------------

__global__ __launch_bounds__(128) void k_logsoftmax(float* __restrict__ out, int F) {
    __shared__ float red[128];
    int row = blockIdx.x;
    int tid = threadIdx.x;
    float v = out[row * F + tid];
    red[tid] = v; __syncthreads();
    for (int o = 64; o > 0; o >>= 1) {
        if (tid < o) red[tid] = fmaxf(red[tid], red[tid + o]);
        __syncthreads();
    }
    float mx = red[0]; __syncthreads();
    red[tid] = expf(v - mx); __syncthreads();
    for (int o = 64; o > 0; o >>= 1) {
        if (tid < o) red[tid] += red[tid + o];
        __syncthreads();
    }
    float lse = mx + logf(red[0]);
    out[row * F + tid] = v - lse;
}

// ---------------------------------------------------------------------------

extern "C" void kernel_launch(void* const* d_in, const int* in_sizes, int n_in,
                              void* d_out, int out_size, void* d_ws, size_t ws_size,
                              hipStream_t stream) {
    const float* x      = (const float*)d_in[0];
    const int*   eidx   = (const int*)  d_in[1];   // [2, E] flat: src then dst
    const float* W1     = (const float*)d_in[2];
    const float* b1     = (const float*)d_in[3];
    const float* gamma1 = (const float*)d_in[4];
    const float* beta1  = (const float*)d_in[5];
    const float* W2     = (const float*)d_in[6];
    const float* b2     = (const float*)d_in[7];
    const float* gamma2 = (const float*)d_in[8];
    const float* beta2  = (const float*)d_in[9];
    const float* W3     = (const float*)d_in[10];
    const float* b3     = (const float*)d_in[11];
    const float* Wskip  = (const float*)d_in[12];
    const float* bskip  = (const float*)d_in[13];

    const int* src = eidx;
    const int* dst = eidx + NE;

    // ---- workspace carve-up (256B aligned) ----
    char* base = (char*)d_ws;
    size_t off = 0;
    auto carve = [&](size_t bytes) -> void* {
        void* p = base + off;
        off += (bytes + 255) & ~(size_t)255;
        return p;
    };
    unsigned short* Xbf   = (unsigned short*)carve((size_t)NN * FIN * 2);
    unsigned short* Midbf = (unsigned short*)carve((size_t)NN * FH * 2);
    unsigned short* W1bf  = (unsigned short*)carve((size_t)FIN * FH * 2);
    unsigned short* W2bf  = (unsigned short*)carve((size_t)FH * FH * 2);
    unsigned short* W3bf  = (unsigned short*)carve((size_t)FH * FOUT * 2);
    unsigned short* Wsbf  = (unsigned short*)carve((size_t)FIN * FOUT * 2);
    float* hbuf   = (float*)carve((size_t)NN * FH * 4);
    float* aggbuf = (float*)carve((size_t)NN * FH * 4);
    float* dinv   = (float*)carve((size_t)NN * 4);
    float* bnsum  = (float*)carve(512 * 4);          // sum[256] + sumsq[256]
    float* bnsq   = bnsum + 256;
    float* mean   = (float*)carve(256 * 4);
    float* rstd   = (float*)carve(256 * 4);
    float* outf   = (float*)d_out;                   // [N, 128]

    const int T = 256;
    auto g1 = [&](long long n) { return (unsigned)((n + T - 1) / T); };

    // ---- degrees -> dinv (in place) ----
    k_zero_f32 <<<g1(NN), T, 0, stream>>>(dinv, NN);
    k_deg_accum<<<g1(NE), T, 0, stream>>>(dst, dinv, NE);
    k_dinv     <<<g1(NN), T, 0, stream>>>(dinv, NN);

    // ---- fp32 -> bf16 conversions ----
    k_cvt_bf16<<<g1((long long)NN * FIN), T, 0, stream>>>(x,     Xbf,  NN * FIN);
    k_cvt_bf16<<<g1(FIN * FH),   T, 0, stream>>>(W1,    W1bf, FIN * FH);
    k_cvt_bf16<<<g1(FH * FH),    T, 0, stream>>>(W2,    W2bf, FH * FH);
    k_cvt_bf16<<<g1(FH * FOUT),  T, 0, stream>>>(W3,    W3bf, FH * FOUT);
    k_cvt_bf16<<<g1(FIN * FOUT), T, 0, stream>>>(Wskip, Wsbf, FIN * FOUT);

    const unsigned GY = (NN / 16 + 3) / 4;   // 782 blocks of 64 rows

    // ---- layer 1: h1 = X @ W1 ; agg ; BN ; ELU -> Midbf ----
    k_gemm_bf16<FIN><<<dim3(FH / 128, GY), 256, 0, stream>>>(Xbf, W1bf, nullptr, hbuf, NN, FH);
    k_self_init<<<g1((long long)NN * FH), T, 0, stream>>>(hbuf, dinv, b1, aggbuf, FH, NN * FH);
    k_scatter  <<<NE, FH, 0, stream>>>(hbuf, src, dst, dinv, aggbuf, FH);
    k_zero_f32 <<<g1(512), T, 0, stream>>>(bnsum, 512);
    k_bn_partial<<<200, 256, 0, stream>>>(aggbuf, bnsum, bnsq, NN, FH, 250);
    k_bn_final  <<<1, 256, 0, stream>>>(bnsum, bnsq, mean, rstd, NN, FH);
    k_bn_elu_bf16<<<g1((long long)NN * FH), T, 0, stream>>>(aggbuf, mean, rstd, gamma1, beta1, Midbf, FH, NN * FH);

    // ---- layer 2: h2 = x1 @ W2 ; agg ; BN ; ELU -> Midbf (reuse) ----
    k_gemm_bf16<FH><<<dim3(FH / 128, GY), 256, 0, stream>>>(Midbf, W2bf, nullptr, hbuf, NN, FH);
    k_self_init<<<g1((long long)NN * FH), T, 0, stream>>>(hbuf, dinv, b2, aggbuf, FH, NN * FH);
    k_scatter  <<<NE, FH, 0, stream>>>(hbuf, src, dst, dinv, aggbuf, FH);
    k_zero_f32 <<<g1(512), T, 0, stream>>>(bnsum, 512);
    k_bn_partial<<<200, 256, 0, stream>>>(aggbuf, bnsum, bnsq, NN, FH, 250);
    k_bn_final  <<<1, 256, 0, stream>>>(bnsum, bnsq, mean, rstd, NN, FH);
    k_bn_elu_bf16<<<g1((long long)NN * FH), T, 0, stream>>>(aggbuf, mean, rstd, gamma2, beta2, Midbf, FH, NN * FH);

    // ---- layer 3: h3 = x2 @ W3 ; skip = X @ Wskip + bskip -> d_out ----
    k_gemm_bf16<FH> <<<dim3(FOUT / 128, GY), 256, 0, stream>>>(Midbf, W3bf, nullptr, hbuf, NN, FOUT);
    k_gemm_bf16<FIN><<<dim3(FOUT / 128, GY), 256, 0, stream>>>(Xbf, Wsbf, bskip, outf, NN, FOUT);
    k_add_self <<<g1((long long)NN * FOUT), T, 0, stream>>>(hbuf, dinv, b3, outf, FOUT, NN * FOUT);
    k_scatter  <<<NE, FOUT, 0, stream>>>(hbuf, src, dst, dinv, outf, FOUT);

    // ---- log_softmax rows, in place ----
    k_logsoftmax<<<NN, FOUT, 0, stream>>>(outf, FOUT);
}
